// DeltaMemory_695784702452
// MI455X (gfx1250) — matrix operand order repaired and checked by reading, not compiled, not run on hardware
//
#include <hip/hip_runtime.h>
#include <math.h>

// ---------------------------------------------------------------------------
// DeltaMemory (gated linear attention) for MI455X / gfx1250, wave32 + WMMA bf16
// B=4, T=2048, D=1024, S=64
// ---------------------------------------------------------------------------

typedef __attribute__((ext_vector_type(16))) __bf16 v16bf;
typedef __attribute__((ext_vector_type(8)))  __bf16 v8bf;
typedef __attribute__((ext_vector_type(8)))  float  v8f;
typedef int v4i_ __attribute__((vector_size(16)));   // matches async-LDS builtin params

#define B_ 4
#define T_ 2048
#define D_ 1024
#define S_ 64

#define AS1 __attribute__((address_space(1)))
#define AS3 __attribute__((address_space(3)))

#if defined(__AMDGCN__) && __has_builtin(__builtin_amdgcn_global_load_async_to_lds_b128)
#define USE_ASYNC_LDS 1
#else
#define USE_ASYNC_LDS 0
#endif

static __device__ __forceinline__ int lane_() { return threadIdx.x & 31; }
static __device__ __forceinline__ int wid_()  { return threadIdx.x >> 5; }

static __device__ __forceinline__ void wait_async0() {
#if defined(__AMDGCN__) && __has_builtin(__builtin_amdgcn_s_wait_asynccnt)
    __builtin_amdgcn_s_wait_asynccnt(0);
#elif defined(__AMDGCN__)
    asm volatile("s_wait_asynccnt 0x0" ::: "memory");
#endif
}

static __device__ __forceinline__ v8f wmma_bf(v16bf a, v16bf b, v8f c) {
    // D = A(16x32 bf16) * B(32x16 bf16) + C(16x16 f32)
    return __builtin_amdgcn_wmma_f32_16x16x32_bf16(
        false, a, false, b, (short)0, c, false, false);
}

static __device__ __forceinline__ v8f v8f_zero() {
    v8f z = {0.f, 0.f, 0.f, 0.f, 0.f, 0.f, 0.f, 0.f};
    return z;
}

// A fragment (16x32, bf16) from row-major f32 matrix: A[m,k] = p[m*ld + k0 + k]
static __device__ __forceinline__ v16bf a_frag_f32(const float* __restrict__ p,
                                                   int ld, int k0) {
    const int l  = lane_();
    const int m  = l & 15;
    const int kh = (l >> 4) * 16;                // K half: {0..15} / {16..31}
    const float* r = p + (size_t)m * ld + k0 + kh;
    v16bf a;
#pragma unroll
    for (int i = 0; i < 8; ++i) { a[i] = (__bf16)r[i]; a[i + 8] = (__bf16)r[8 + i]; }
    return a;
}

// A fragment (16x32, bf16) from row-major bf16 matrix
static __device__ __forceinline__ v16bf a_frag_bf(const __bf16* __restrict__ p,
                                                  int ld, int k0) {
    const int l  = lane_();
    const int m  = l & 15;
    const int kh = (l >> 4) * 16;
    const __bf16* r = p + (size_t)m * ld + k0 + kh;
    v16bf a;
#pragma unroll
    for (int i = 0; i < 8; ++i) { a[i] = r[i]; a[i + 8] = r[8 + i]; }
    return a;
}

// B fragment (32x16) where B = P^T of a row-major f32 matrix:
// B[k,n] = P[(n0+n)*ld + k0 + k]  (contiguous along k)
static __device__ __forceinline__ v16bf b_frag_rowsT_f32(const float* __restrict__ p,
                                                         int ld, int n0, int k0) {
    const int l  = lane_();
    const int n  = n0 + (l & 15);
    const int kb = k0 + (l >> 4) * 16;
    const float* r = p + (size_t)n * ld + kb;
    v16bf b;
#pragma unroll
    for (int i = 0; i < 16; ++i) b[i] = (__bf16)r[i];
    return b;
}

// Same, from bf16 source (32 contiguous bytes per lane)
static __device__ __forceinline__ v16bf b_frag_rowsT_bf(const __bf16* __restrict__ p,
                                                        int ld, int n0, int k0) {
    const int l  = lane_();
    const int n  = n0 + (l & 15);
    const int kb = k0 + (l >> 4) * 16;
    const __bf16* r = p + (size_t)n * ld + kb;
    v16bf b;
#pragma unroll
    for (int i = 0; i < 16; ++i) b[i] = r[i];
    return b;
}

// B fragment with only column 0 populated from a vector: B[k,0] = wg[k0+k]
static __device__ __forceinline__ v16bf b_frag_vec_f32(const float* __restrict__ wg,
                                                       int k0) {
    const int l  = lane_();
    const int n  = l & 15;
    const int kb = k0 + (l >> 4) * 16;
    v16bf b;
    if (n == 0) {
#pragma unroll
        for (int i = 0; i < 16; ++i) b[i] = (__bf16)wg[kb + i];
    } else {
#pragma unroll
        for (int i = 0; i < 16; ++i) b[i] = (__bf16)0.f;
    }
    return b;
}

// B fragment (32x16, bf16) from row-major bf16 matrix: B[k,n] = p[(k0+k)*ld + n0+n]
static __device__ __forceinline__ v16bf b_frag_rm_bf(const __bf16* __restrict__ p,
                                                     int ld, int k0, int n0) {
    const int l  = lane_();
    const int n  = n0 + (l & 15);
    const int kb = k0 + (l >> 4) * 16;
    v16bf b;
#pragma unroll
    for (int i = 0; i < 16; ++i) b[i] = p[(size_t)(kb + i) * ld + n];
    return b;
}

// ---------------------------------------------------------------------------
// K0: one-time f32 -> bf16 weight conversion
// ---------------------------------------------------------------------------
__global__ __launch_bounds__(256) void k_cvt(const float* __restrict__ src,
                                             __bf16* __restrict__ dst, int n)
{
    const int i = (blockIdx.x * 256 + threadIdx.x) * 4;
    if (i + 3 < n) {
#pragma unroll
        for (int j = 0; j < 4; ++j) dst[i + j] = (__bf16)src[i + j];
    }
}

// ---------------------------------------------------------------------------
// K1: projections k/v/q + alpha (all WMMA), RoPE, gates.  128 t-rows / block.
// ---------------------------------------------------------------------------
__global__ __launch_bounds__(256) void k_proj(
    const float* __restrict__ x,
    const __bf16* __restrict__ wkb, const __bf16* __restrict__ wvb,
    const __bf16* __restrict__ wqb, const float* __restrict__ Wg,
    const float* __restrict__ bg,
    __bf16* __restrict__ krb, __bf16* __restrict__ qrb,
    __bf16* __restrict__ avb, float* __restrict__ logg)
{
    __shared__ float s_alpha[128];
    const int b  = blockIdx.x / (T_ / 128);
    const int t0 = (blockIdx.x % (T_ / 128)) * 128;
    const int w  = wid_(), l = lane_();
    const int rbase = (l >> 4) * 8, c = l & 15;
    const float bgv = bg[0];
    const float* xw = x + ((size_t)b * T_ + t0 + w * 16) * D_;

    // --- fused GEMMs: k/v/q (N=64 each) + alpha column (N=1), K=1024 ---
    v8f acc[3][4], accA;
#pragma unroll
    for (int j = 0; j < 3; ++j)
#pragma unroll
        for (int nt = 0; nt < 4; ++nt) acc[j][nt] = v8f_zero();
    accA = v8f_zero();

    for (int k0 = 0; k0 < D_; k0 += 32) {
        v16bf a = a_frag_f32(xw, D_, k0);
#pragma unroll
        for (int nt = 0; nt < 4; ++nt) {
            acc[0][nt] = wmma_bf(a, b_frag_rowsT_bf(wkb, D_, nt * 16, k0), acc[0][nt]);
            acc[1][nt] = wmma_bf(a, b_frag_rowsT_bf(wvb, D_, nt * 16, k0), acc[1][nt]);
            acc[2][nt] = wmma_bf(a, b_frag_rowsT_bf(wqb, D_, nt * 16, k0), acc[2][nt]);
        }
        accA = wmma_bf(a, b_frag_vec_f32(Wg, k0), accA);
    }

    // alpha lives in column 0 of accA (lanes 0 and 16)
    if (c == 0) {
#pragma unroll
        for (int r = 0; r < 8; ++r)
            s_alpha[w * 16 + rbase + r] = 1.f / (1.f + __expf(-(accA[r] + bgv)));
    }
    __syncthreads();

    // --- epilogue: RoPE (pairs s <-> s+32), |k|^2, gate, alpha*v, stores ---
    const float LN1E4 = 9.210340371976184f;  // ln(10000)
#pragma unroll
    for (int nt = 0; nt < 2; ++nt) {
        const int s = nt * 16 + c;                        // 0..31
        const float invf = __expf(-(float)s * (1.f / 32.f) * LN1E4);
#pragma unroll
        for (int r = 0; r < 8; ++r) {
            const int t = t0 + w * 16 + rbase + r;
            const float f  = (float)t * invf;
            const float cs = __cosf(f), sn = __sinf(f);
            const float k1 = acc[0][nt][r], k2 = acc[0][nt + 2][r];
            acc[0][nt][r]     = k1 * cs - k2 * sn;
            acc[0][nt + 2][r] = k2 * cs + k1 * sn;
            const float q1 = acc[2][nt][r], q2 = acc[2][nt + 2][r];
            acc[2][nt][r]     = q1 * cs - q2 * sn;
            acc[2][nt + 2][r] = q2 * cs + q1 * sn;
        }
    }

    float knorm[8];
#pragma unroll
    for (int r = 0; r < 8; ++r) {
        float p = 0.f;
#pragma unroll
        for (int nt = 0; nt < 4; ++nt) p += acc[0][nt][r] * acc[0][nt][r];
#pragma unroll
        for (int m = 8; m >= 1; m >>= 1) p += __shfl_xor(p, m, 32);  // 16-lane groups
        knorm[r] = p;
    }

    const size_t base = ((size_t)b * T_ + t0 + w * 16) * S_;
#pragma unroll
    for (int r = 0; r < 8; ++r) {
        const int localrow = w * 16 + rbase + r;
        const int t = t0 + localrow;
        const float alpha = s_alpha[localrow];
        const float z = alpha * knorm[r] * 0.1f;
        const float g = 1.f / (1.f + __expf(z));     // exp(-softplus(z))
        if (c == 0) logg[(size_t)b * T_ + t] = __logf(g + 1e-8f);
        const size_t rowoff = base + (size_t)(rbase + r) * S_;
#pragma unroll
        for (int nt = 0; nt < 4; ++nt) {
            const int col = nt * 16 + c;
            krb[rowoff + col] = (__bf16)acc[0][nt][r];
            qrb[rowoff + col] = (__bf16)acc[2][nt][r];
            avb[rowoff + col] = (__bf16)(alpha * acc[1][nt][r]);
        }
    }
}

// ---------------------------------------------------------------------------
// K2: per-batch inclusive cumsum of log_g over T
// ---------------------------------------------------------------------------
__global__ __launch_bounds__(256) void k_scan(const float* __restrict__ logg,
                                              float* __restrict__ cum)
{
    __shared__ float s_sum[256];
    const int b = blockIdx.x, tid = threadIdx.x;
    const float* in = logg + (size_t)b * T_;
    float* out = cum + (size_t)b * T_;

    float v[8], run = 0.f;
#pragma unroll
    for (int i = 0; i < 8; ++i) { run += in[tid * 8 + i]; v[i] = run; }
    s_sum[tid] = run;
    __syncthreads();
    float off = 0.f;
    for (int i = 0; i < tid; ++i) off += s_sum[i];
#pragma unroll
    for (int i = 0; i < 8; ++i) out[tid * 8 + i] = v[i] + off;
}

// ---------------------------------------------------------------------------
// K3 helper: cooperative staging of one 32-key block.
// kr tile is staged row-major (async global->LDS when available);
// alpha*v tile is staged transposed (avT[d][s]) so intra-GEMM B-fragments are
// 32 contiguous bytes per lane.
// ---------------------------------------------------------------------------
static __device__ __forceinline__ void stage_kv(const __bf16* __restrict__ kb,
                                                const __bf16* __restrict__ ab,
                                                int s0, __bf16* krT, __bf16* avT)
{
    const int tid = threadIdx.x;
    const int row = tid >> 3;            // 0..31 (key row)
    const int ch  = (tid & 7) * 8;       // 8 bf16 (16 B) per thread
    const __bf16* gk = kb + (size_t)(s0 + row) * S_ + ch;
    __bf16* lk = krT + row * S_ + ch;
#if USE_ASYNC_LDS
    __builtin_amdgcn_global_load_async_to_lds_b128(
        (AS1 v4i_*)gk, (AS3 v4i_*)lk, 0, 0);
#else
    *(v8bf*)lk = *(const v8bf*)gk;
#endif
    const v8bf av = *(const v8bf*)(ab + (size_t)(s0 + row) * S_ + ch);
#pragma unroll
    for (int j = 0; j < 8; ++j) avT[(ch + j) * 32 + row] = av[j];
}

// ---------------------------------------------------------------------------
// K3: causal decayed attention (intra) + state read-out (past) -> retrieved
// Block-uniform key loop with double-buffered LDS tiles.
// ---------------------------------------------------------------------------
__global__ __launch_bounds__(256) void k_attn(
    const __bf16* __restrict__ krb, const __bf16* __restrict__ qrb,
    const __bf16* __restrict__ avb, const float* __restrict__ cum,
    const float* __restrict__ state, __bf16* __restrict__ retb)
{
    __shared__ float  s_cum[T_];            // 8 KB
    __shared__ __bf16 s_sc[8][16 * 32];     // per-wave decayed-score tile, 8 KB
    __shared__ __bf16 s_kr[2][32 * S_];     // key tiles (row-major), 8 KB
    __shared__ __bf16 s_avT[2][S_ * 32];    // alpha*v tiles (transposed), 8 KB
    const int b  = blockIdx.x / (T_ / 128);
    const int t0 = (blockIdx.x % (T_ / 128)) * 128;
    const int w  = wid_(), l = lane_();

    const float* cb = cum + (size_t)b * T_;
    for (int i = threadIdx.x; i < T_; i += 256) s_cum[i] = cb[i];

    const int qt = t0 + w * 16;
    const __bf16* qp = qrb + ((size_t)b * T_ + qt) * S_;
    const v16bf aq0 = a_frag_bf(qp, S_, 0);
    const v16bf aq1 = a_frag_bf(qp, S_, 32);

    const __bf16* kb = krb + (size_t)b * T_ * S_;
    const __bf16* ab = avb + (size_t)b * T_ * S_;
    stage_kv(kb, ab, 0, s_kr[0], s_avT[0]);
#if USE_ASYNC_LDS
    wait_async0();
#endif
    __syncthreads();

    const int rbase = (l >> 4) * 8, c = l & 15;
    float csh[8], dct[8];
#pragma unroll
    for (int r = 0; r < 8; ++r) {
        const int t = qt + rbase + r;
        csh[r] = (t == 0) ? 0.f : s_cum[t - 1];   // cum_shift[t]
        dct[r] = __expf(csh[r]);                  // decay_t[t]
    }

    // past = (q_r @ state^T) * decay_t  seeds the accumulator
    const float* st = state + (size_t)b * S_ * S_;
    v8f acc[4];
#pragma unroll
    for (int nt = 0; nt < 4; ++nt) {
        v8f z = v8f_zero();
        z = wmma_bf(aq0, b_frag_rowsT_f32(st, S_, nt * 16, 0),  z);
        z = wmma_bf(aq1, b_frag_rowsT_f32(st, S_, nt * 16, 32), z);
#pragma unroll
        for (int r = 0; r < 8; ++r) z[r] *= dct[r];
        acc[nt] = z;
    }

    __bf16* my = &s_sc[w][0];
    const int nIter = (t0 + 126) / 32 + 1;     // keys [0, t0+127) covered

    for (int it = 0; it < nIter; ++it) {
        const int s0 = it * 32;
        const __bf16* krT = s_kr[it & 1];
        const __bf16* avT = s_avT[it & 1];
        if (it + 1 < nIter)
            stage_kv(kb, ab, s0 + 32, s_kr[(it + 1) & 1], s_avT[(it + 1) & 1]);

        // scores[t, s0..s0+31] = q_r . k_r^T   (K = 64), from LDS key tile
        v8f sc[2];
#pragma unroll
        for (int kt = 0; kt < 2; ++kt) {
            v8f z = v8f_zero();
            z = wmma_bf(aq0, b_frag_rowsT_bf(krT, S_, kt * 16, 0),  z);
            z = wmma_bf(aq1, b_frag_rowsT_bf(krT, S_, kt * 16, 32), z);
            sc[kt] = z;
        }
        // decay + causal mask, round to bf16 in per-wave LDS tile (16x32)
#pragma unroll
        for (int kt = 0; kt < 2; ++kt) {
            const int s = s0 + kt * 16 + c;
            const float cs = s_cum[s];
#pragma unroll
            for (int r = 0; r < 8; ++r) {
                const int t = qt + rbase + r;
                float v = 0.f;
                if (s < t) v = sc[kt][r] * __expf(csh[r] - cs);
                my[(rbase + r) * 32 + kt * 16 + c] = (__bf16)v;
            }
        }
        // wave-private LDS round-trip: wait on this wave's DS ops only
        asm volatile("s_wait_dscnt 0x0" ::: "memory");
        const v16bf as = a_frag_bf(my, 32, 0);
#pragma unroll
        for (int nt = 0; nt < 4; ++nt)
            acc[nt] = wmma_bf(as, b_frag_rowsT_bf(avT, 32, nt * 16, 0), acc[nt]);

#if USE_ASYNC_LDS
        wait_async0();        // staging for next buffer complete (this wave)
#endif
        __syncthreads();      // everyone done reading current buffers
    }

    __bf16* rp = retb + ((size_t)b * T_ + qt) * S_;
#pragma unroll
    for (int nt = 0; nt < 4; ++nt)
#pragma unroll
        for (int r = 0; r < 8; ++r)
            rp[(size_t)(rbase + r) * S_ + nt * 16 + c] = (__bf16)acc[nt][r];
}

// ---------------------------------------------------------------------------
// K4: new_state = decayed_v^T @ k_r + state * final_decay   (64x2048x64 / batch)
// ---------------------------------------------------------------------------
__global__ __launch_bounds__(128) void k_state(
    const __bf16* __restrict__ krb, const __bf16* __restrict__ avb,
    const float* __restrict__ cum, const float* __restrict__ state,
    float* __restrict__ state_out)
{
    __shared__ float s_cum[T_];
    const int b = blockIdx.x, w = wid_(), l = lane_();
    const float* cb = cum + (size_t)b * T_;
    for (int i = threadIdx.x; i < T_; i += 128) s_cum[i] = cb[i];
    __syncthreads();

    const float fin = s_cum[T_ - 1];
    const int m0 = w * 16;
    const int mrow = l & 15;
    const int kh   = (l >> 4) * 16;
    const __bf16* ab = avb + (size_t)b * T_ * S_;
    const __bf16* kb = krb + (size_t)b * T_ * S_;

    v8f acc[4];
#pragma unroll
    for (int nt = 0; nt < 4; ++nt) acc[nt] = v8f_zero();

    for (int t0 = 0; t0 < T_; t0 += 32) {
        // A[m,k] = av[t0+k, m0+m] * exp(fin - cum[t0+k])   (decayed_v transposed)
        v16bf a;
#pragma unroll
        for (int i = 0; i < 8; ++i) {
            const int ta = t0 + kh + i;
            const int tb = t0 + 8 + kh + i;
            a[i]     = (__bf16)((float)ab[(size_t)ta * S_ + m0 + mrow] * __expf(fin - s_cum[ta]));
            a[i + 8] = (__bf16)((float)ab[(size_t)tb * S_ + m0 + mrow] * __expf(fin - s_cum[tb]));
        }
#pragma unroll
        for (int nt = 0; nt < 4; ++nt)
            acc[nt] = wmma_bf(a, b_frag_rm_bf(kb, S_, t0, nt * 16), acc[nt]);
    }

    const float fd = __expf(fin);
    const int rbase = (l >> 4) * 8, c = l & 15;
    const float* st = state + (size_t)b * S_ * S_;
    float* so = state_out + (size_t)b * S_ * S_;
#pragma unroll
    for (int nt = 0; nt < 4; ++nt)
#pragma unroll
        for (int r = 0; r < 8; ++r) {
            const int row = m0 + rbase + r, col = nt * 16 + c;
            so[(size_t)row * S_ + col] = st[(size_t)row * S_ + col] * fd + acc[nt][r];
        }
}

// ---------------------------------------------------------------------------
// K5: outputs = retrieved @ Wout^T + b_out    (8192 x 64 -> 1024)
// ---------------------------------------------------------------------------
__global__ __launch_bounds__(256) void k_out(
    const __bf16* __restrict__ retb, const __bf16* __restrict__ woutb,
    const float* __restrict__ bout, float* __restrict__ out)
{
    const int b  = blockIdx.x / (T_ / 128);
    const int t0 = (blockIdx.x % (T_ / 128)) * 128;
    const int n0 = blockIdx.y * 64;
    const int w  = wid_(), l = lane_();

    const __bf16* rp = retb + ((size_t)b * T_ + t0 + w * 16) * S_;
    const v16bf a0 = a_frag_bf(rp, S_, 0);
    const v16bf a1 = a_frag_bf(rp, S_, 32);

    v8f acc[4];
#pragma unroll
    for (int nt = 0; nt < 4; ++nt) {
        v8f z = v8f_zero();
        z = wmma_bf(a0, b_frag_rowsT_bf(woutb, S_, n0 + nt * 16, 0),  z);
        z = wmma_bf(a1, b_frag_rowsT_bf(woutb, S_, n0 + nt * 16, 32), z);
        acc[nt] = z;
    }

    const int rbase = (l >> 4) * 8, c = l & 15;
    float* op = out + ((size_t)b * T_ + t0 + w * 16) * D_;
#pragma unroll
    for (int nt = 0; nt < 4; ++nt) {
        const int col = n0 + nt * 16 + c;
        const float bb = bout[col];
#pragma unroll
        for (int r = 0; r < 8; ++r)
            op[(size_t)(rbase + r) * D_ + col] = acc[nt][r] + bb;
    }
}

// ---------------------------------------------------------------------------
extern "C" void kernel_launch(void* const* d_in, const int* in_sizes, int n_in,
                              void* d_out, int out_size, void* d_ws, size_t ws_size,
                              hipStream_t stream)
{
    (void)in_sizes; (void)n_in; (void)out_size; (void)ws_size;
    const float* x     = (const float*)d_in[0];
    const float* state = (const float*)d_in[1];
    const float* Wk    = (const float*)d_in[2];
    const float* Wv    = (const float*)d_in[3];
    const float* Wq    = (const float*)d_in[4];
    const float* Wout  = (const float*)d_in[5];
    const float* bout  = (const float*)d_in[6];
    const float* Wg    = (const float*)d_in[7];
    const float* bg    = (const float*)d_in[8];

    float* out       = (float*)d_out;                  // B*T*D
    float* state_out = out + (size_t)B_ * T_ * D_;     // B*S*S

    char* ws = (char*)d_ws;
    size_t o = 0;
    auto carve = [&](size_t bytes) -> void* {
        void* p = (void*)(ws + o);
        o += (bytes + 255) & ~(size_t)255;
        return p;
    };
    float*  logg  = (float*)carve(sizeof(float) * (size_t)B_ * T_);
    float*  cum   = (float*)carve(sizeof(float) * (size_t)B_ * T_);
    __bf16* krb   = (__bf16*)carve(2 * (size_t)B_ * T_ * S_);
    __bf16* qrb   = (__bf16*)carve(2 * (size_t)B_ * T_ * S_);
    __bf16* avb   = (__bf16*)carve(2 * (size_t)B_ * T_ * S_);
    __bf16* retb  = (__bf16*)carve(2 * (size_t)B_ * T_ * S_);
    __bf16* wkb   = (__bf16*)carve(2 * (size_t)S_ * D_);
    __bf16* wvb   = (__bf16*)carve(2 * (size_t)S_ * D_);
    __bf16* wqb   = (__bf16*)carve(2 * (size_t)S_ * D_);
    __bf16* woutb = (__bf16*)carve(2 * (size_t)D_ * S_);

    const int nw = S_ * D_;                            // 65536
    k_cvt<<<dim3(nw / (256 * 4)), dim3(256), 0, stream>>>(Wk,   wkb,   nw);
    k_cvt<<<dim3(nw / (256 * 4)), dim3(256), 0, stream>>>(Wv,   wvb,   nw);
    k_cvt<<<dim3(nw / (256 * 4)), dim3(256), 0, stream>>>(Wq,   wqb,   nw);
    k_cvt<<<dim3(nw / (256 * 4)), dim3(256), 0, stream>>>(Wout, woutb, nw);

    k_proj <<<dim3(B_ * T_ / 128), dim3(256), 0, stream>>>(x, wkb, wvb, wqb, Wg, bg,
                                                           krb, qrb, avb, logg);
    k_scan <<<dim3(B_),            dim3(256), 0, stream>>>(logg, cum);
    k_attn <<<dim3(B_ * T_ / 128), dim3(256), 0, stream>>>(krb, qrb, avb, cum, state, retb);
    k_state<<<dim3(B_),            dim3(128), 0, stream>>>(krb, avb, cum, state, state_out);
    k_out  <<<dim3(B_ * T_ / 128, D_ / 64), dim3(256), 0, stream>>>(retb, woutb, bout, out);
}
